// MultiHeadAttention_27410481283187
// MI455X (gfx1250) — matrix operand order repaired and checked by reading, not compiled, not run on hardware
//
#include <hip/hip_runtime.h>
#include <hip/hip_bf16.h>

typedef __attribute__((ext_vector_type(16))) _Float16 v16h;
typedef __attribute__((ext_vector_type(8)))  float    v8f;

#define WMMA_F16(a, b, c) \
    __builtin_amdgcn_wmma_f32_16x16x32_f16(false, (a), false, (b), (short)0, (c), false, false)

// ---------------------------------------------------------------------------
// DPP16 butterfly reduction over each 16-lane row (pure VALU, no LDS, no waits)
// ---------------------------------------------------------------------------
template <int CTRL>
__device__ __forceinline__ float dpp_permf(float v) {
    int r = __builtin_amdgcn_update_dpp(0, __builtin_bit_cast(int, v),
                                        CTRL, 0xf, 0xf, true);
    return __builtin_bit_cast(float, r);
}
__device__ __forceinline__ float row16_max(float v) {
    v = fmaxf(v, dpp_permf<0xB1>(v));    // quad_perm [1,0,3,2]
    v = fmaxf(v, dpp_permf<0x4E>(v));    // quad_perm [2,3,0,1]
    v = fmaxf(v, dpp_permf<0x141>(v));   // row_half_mirror
    v = fmaxf(v, dpp_permf<0x140>(v));   // row_mirror
    return v;
}
__device__ __forceinline__ float row16_sum(float v) {
    v += dpp_permf<0xB1>(v);
    v += dpp_permf<0x4E>(v);
    v += dpp_permf<0x141>(v);
    v += dpp_permf<0x140>(v);
    return v;
}

// ---------------------------------------------------------------------------
// Fragment loaders (wave32). Source is row-major with element stride `ld`.
// A fragment (16x32, MxK): row = lane&15; elems 0..7 = K 8*half+{0..7},
//                          elems 8..15 = K 16+8*half+{0..7}   (ISA 7.12.2)
// B fragment (32x16, KxN) loaded from N-major (B^T) storage: row = n = lane&15;
//                          elems 0..7 = K 8*half+{0..7}, elems 8..15 = K 16+8*half+{0..7}
// ---------------------------------------------------------------------------
__device__ __forceinline__ v16h load_frag_a(const _Float16* __restrict__ base,
                                            int ld, int lane) {
    int row = lane & 15, half = lane >> 4;
    union { v16h v; uint4 q[2]; } u;
    const _Float16* p = base + (size_t)row * ld + 8 * half;
    u.q[0] = *(const uint4*)(p);
    u.q[1] = *(const uint4*)(p + 16);
    return u.v;
}

__device__ __forceinline__ v16h load_frag_b(const _Float16* __restrict__ base,
                                            int ld, int lane) {
    int row = lane & 15, half = lane >> 4;
    union { v16h v; uint4 q[2]; } u;
    const _Float16* p = base + (size_t)row * ld + 8 * half;
    u.q[0] = *(const uint4*)(p);
    u.q[1] = *(const uint4*)(p + 16);
    return u.v;
}

__device__ __forceinline__ v16h load_frag_a_lds(const _Float16* base,
                                                int ld, int lane) {
    int row = lane & 15, half = lane >> 4;
    union { v16h v; uint4 q[2]; } u;
    const _Float16* p = base + row * ld + 8 * half;
    u.q[0] = *(const uint4*)(p);
    u.q[1] = *(const uint4*)(p + 16);
    return u.v;
}

// ---------------------------------------------------------------------------
// Prep kernels
// ---------------------------------------------------------------------------
__global__ void cvt_f32_to_f16(const float* __restrict__ in,
                               _Float16* __restrict__ out, int n) {
    int i = blockIdx.x * blockDim.x + threadIdx.x;
    if (i < n) out[i] = (_Float16)in[i];
}

// in: 1024x1024 (k-major rows), out: W^T as 1024x1024 (n-major rows), f16
__global__ void transpose_w_f16(const float* __restrict__ in,
                                _Float16* __restrict__ out) {
    int i = blockIdx.x * blockDim.x + threadIdx.x;   // n*1024 + k
    int n = i >> 10, k = i & 1023;
    out[i] = (_Float16)in[k * 1024 + n];
}

// ---------------------------------------------------------------------------
// WMMA GEMM: Y[M=8192, N=1024] = A(f16, row-major, ld=1024) @ BT^T + bias
// Block: 256 thr = 8 waves, wave tile 32x64, block tile 128x128.
// MODE 0: f16 [B,H,T,D] (q,k); MODE 1: f16 [B,H,D,T] (v^T); MODE 2: f32 [B,T,C]
// ---------------------------------------------------------------------------
template <int MODE>
__global__ __launch_bounds__(256) void gemm_wmma(
    const _Float16* __restrict__ A, const _Float16* __restrict__ BT,
    const float* __restrict__ bias, void* __restrict__ out) {
    const int C = 1024;
    int lane = threadIdx.x & 31;
    int w    = threadIdx.x >> 5;      // 0..7
    int wm = w >> 1, wn = w & 1;
    int m0 = blockIdx.y * 128 + wm * 32;
    int n0 = blockIdx.x * 128 + wn * 64;

    v8f c[2][4];
#pragma unroll
    for (int i = 0; i < 2; ++i)
#pragma unroll
        for (int j = 0; j < 4; ++j) c[i][j] = (v8f){};

    for (int k0 = 0; k0 < C; k0 += 32) {
        v16h af[2], bf[4];
#pragma unroll
        for (int i = 0; i < 2; ++i)
            af[i] = load_frag_a(A + (size_t)(m0 + 16 * i) * C + k0, C, lane);
#pragma unroll
        for (int j = 0; j < 4; ++j)
            bf[j] = load_frag_b(BT + (size_t)(n0 + 16 * j) * C + k0, C, lane);
#pragma unroll
        for (int i = 0; i < 2; ++i)
#pragma unroll
            for (int j = 0; j < 4; ++j) c[i][j] = WMMA_F16(af[i], bf[j], c[i][j]);
    }

    int half = lane >> 4, col = lane & 15;
#pragma unroll
    for (int i = 0; i < 2; ++i)
#pragma unroll
        for (int j = 0; j < 4; ++j) {
            int n = n0 + 16 * j + col;
            float bv = bias[n];
#pragma unroll
            for (int r = 0; r < 8; ++r) {
                int m = m0 + 16 * i + r + 8 * half;
                float val = c[i][j][r] + bv;
                if (MODE == 2) {
                    ((float*)out)[(size_t)m * C + n] = val;
                } else {
                    int b = m >> 11, t = m & 2047;
                    int h = n >> 6,  d = n & 63;
                    size_t idx = (MODE == 0)
                        ? ((size_t)(b * 16 + h) * 2048 + t) * 64 + d
                        : ((size_t)(b * 16 + h) * 64 + d) * 2048 + t;
                    ((_Float16*)out)[idx] = (_Float16)val;
                }
            }
        }
}

// ---------------------------------------------------------------------------
// Flash attention. One block = 4 waves = 64 query rows for one (b,h).
// Software-pipelined: all 16 K/V fragments issued up front each chunk so
// V-load latency hides behind softmax VALU and K-load latency behind WMMAs.
// ---------------------------------------------------------------------------
__global__ __launch_bounds__(128) void attn_wmma(
    const _Float16* __restrict__ Q, const _Float16* __restrict__ K,
    const _Float16* __restrict__ VT, _Float16* __restrict__ Aout) {
    const int T = 2048, D = 64;
    __shared__ _Float16 sP[4][16 * 64];     // per-wave P staging (2 KB each)

    int lane = threadIdx.x & 31;
    int w    = threadIdx.x >> 5;            // 0..3
    int half = lane >> 4;
    int bh   = blockIdx.y;                  // 0..63
    int b = bh >> 4, h = bh & 15;
    int q0 = blockIdx.x * 64 + w * 16;      // this wave's query rows

    const _Float16* qh = Q  + (size_t)bh * T * D;
    const _Float16* kh = K  + (size_t)bh * T * D;
    const _Float16* vh = VT + (size_t)bh * D * T;

    v16h qa[2];
    qa[0] = load_frag_a(qh + (size_t)q0 * D +  0, D, lane);
    qa[1] = load_frag_a(qh + (size_t)q0 * D + 32, D, lane);
#pragma unroll
    for (int e = 0; e < 16; ++e) {          // fold 1/sqrt(64) into q once
        qa[0][e] = qa[0][e] * (_Float16)0.125f;
        qa[1][e] = qa[1][e] * (_Float16)0.125f;
    }

    v8f o[4];
    float mrun[8], lrun[8];
#pragma unroll
    for (int j = 0; j < 4; ++j) o[j] = (v8f){};
#pragma unroll
    for (int r = 0; r < 8; ++r) { mrun[r] = -1e30f; lrun[r] = 0.0f; }

    for (int s0 = 0; s0 < T; s0 += 64) {
        // ---- issue ALL fragment loads for this chunk (distinct registers) ----
        v16h kb[4][2], vb[4][2];
#pragma unroll
        for (int j = 0; j < 4; ++j) {
            kb[j][0] = load_frag_b(kh + (size_t)(s0 + 16 * j) * D +  0, D, lane);
            kb[j][1] = load_frag_b(kh + (size_t)(s0 + 16 * j) * D + 32, D, lane);
        }
#pragma unroll
        for (int j = 0; j < 4; ++j) {
            vb[j][0] = load_frag_b(vh + (size_t)(16 * j) * T + s0 +  0, T, lane);
            vb[j][1] = load_frag_b(vh + (size_t)(16 * j) * T + s0 + 32, T, lane);
        }
        // prefetch next chunk's K/V lines (global_prefetch_b8, no counters)
        if (s0 + 64 < T) {
            __builtin_prefetch(kh + (size_t)(s0 + 64) * D + lane * 128, 0, 1);
            __builtin_prefetch(vh + (size_t)lane * T        + s0 + 64, 0, 1);
            __builtin_prefetch(vh + (size_t)(lane + 32) * T + s0 + 64, 0, 1);
        }
        // ---- S = (q*scale) @ k^T : 8 WMMAs, staged loadcnt waits ----
        v8f s[4];
#pragma unroll
        for (int j = 0; j < 4; ++j) {
            s[j] = (v8f){};
            s[j] = WMMA_F16(qa[0], kb[j][0], s[j]);
            s[j] = WMMA_F16(qa[1], kb[j][1], s[j]);
        }
        // ---- online softmax (DPP16 reductions) overlaps V loads in flight ----
#pragma unroll
        for (int r = 0; r < 8; ++r) {
            float rmax = fmaxf(fmaxf(s[0][r], s[1][r]), fmaxf(s[2][r], s[3][r]));
            rmax = row16_max(rmax);
            float mnew = fmaxf(mrun[r], rmax);
            float cf   = __expf(mrun[r] - mnew);
            mrun[r] = mnew;
            float rsum = 0.0f;
#pragma unroll
            for (int j = 0; j < 4; ++j) {
                float p = __expf(s[j][r] - mnew);
                s[j][r] = p;
                rsum += p;
            }
            rsum = row16_sum(rsum);
            lrun[r] = lrun[r] * cf + rsum;
#pragma unroll
            for (int j = 0; j < 4; ++j) o[j][r] *= cf;
        }
        // ---- stage P (16x64) through per-wave LDS, reload as A fragments ----
        _Float16* sp = sP[w];
#pragma unroll
        for (int j = 0; j < 4; ++j)
#pragma unroll
            for (int r = 0; r < 8; ++r) {
                int m = r + 8 * half, cn = 16 * j + (lane & 15);
                sp[m * 64 + cn] = (_Float16)s[j][r];
            }
        v16h pa0 = load_frag_a_lds(sp +  0, 64, lane);
        v16h pa1 = load_frag_a_lds(sp + 32, 64, lane);
        // ---- O += P @ V ----
#pragma unroll
        for (int j = 0; j < 4; ++j) {
            o[j] = WMMA_F16(pa0, vb[j][0], o[j]);
            o[j] = WMMA_F16(pa1, vb[j][1], o[j]);
        }
    }
    // ---- normalize rows, scatter to [B,T,C] f16 ----
#pragma unroll
    for (int r = 0; r < 8; ++r) {
        float inv = 1.0f / lrun[r];
        int t = q0 + r + 8 * half;
#pragma unroll
        for (int j = 0; j < 4; ++j) {
            int d = 16 * j + (lane & 15);
            Aout[((size_t)b * 2048 + t) * 1024 + h * 64 + d] =
                (_Float16)(o[j][r] * inv);
        }
    }
}

// ---------------------------------------------------------------------------
extern "C" void kernel_launch(void* const* d_in, const int* in_sizes, int n_in,
                              void* d_out, int out_size, void* d_ws, size_t ws_size,
                              hipStream_t stream) {
    const float* x  = (const float*)d_in[0];
    const float* Wq = (const float*)d_in[1];
    const float* bq = (const float*)d_in[2];
    const float* Wk = (const float*)d_in[3];
    const float* bk = (const float*)d_in[4];
    const float* Wv = (const float*)d_in[5];
    const float* bv = (const float*)d_in[6];
    const float* Wp = (const float*)d_in[7];
    const float* bp = (const float*)d_in[8];

    const size_t M  = 4 * 2048;          // B*T
    const size_t C  = 1024;
    const size_t NX = M * C;             // 8,388,608

    char* ws = (char*)d_ws;
    size_t off = 0;
    _Float16* xh   = (_Float16*)(ws + off); off += NX * 2;        // x f16
    _Float16* WqT  = (_Float16*)(ws + off); off += C * C * 2;
    _Float16* WkT  = (_Float16*)(ws + off); off += C * C * 2;
    _Float16* WvT  = (_Float16*)(ws + off); off += C * C * 2;
    _Float16* WpT  = (_Float16*)(ws + off); off += C * C * 2;
    _Float16* qb   = (_Float16*)(ws + off); off += NX * 2;        // [B,H,T,D]
    _Float16* kb   = (_Float16*)(ws + off); off += NX * 2;        // [B,H,T,D]
    _Float16* vT   = (_Float16*)(ws + off); off += NX * 2;        // [B,H,D,T]
    _Float16* aout = (_Float16*)(ws + off); off += NX * 2;        // [B,T,C]

    cvt_f32_to_f16<<<(int)(NX / 256), 256, 0, stream>>>(x, xh, (int)NX);
    transpose_w_f16<<<(int)(C * C / 256), 256, 0, stream>>>(Wq, WqT);
    transpose_w_f16<<<(int)(C * C / 256), 256, 0, stream>>>(Wk, WkT);
    transpose_w_f16<<<(int)(C * C / 256), 256, 0, stream>>>(Wv, WvT);
    transpose_w_f16<<<(int)(C * C / 256), 256, 0, stream>>>(Wp, WpT);

    dim3 ggrid(1024 / 128, (unsigned)(M / 128));   // (8, 64)
    gemm_wmma<0><<<ggrid, 256, 0, stream>>>(xh, WqT, bq, qb);
    gemm_wmma<0><<<ggrid, 256, 0, stream>>>(xh, WkT, bk, kb);
    gemm_wmma<1><<<ggrid, 256, 0, stream>>>(xh, WvT, bv, vT);

    dim3 agrid(2048 / 64, 64);                     // (32, B*H)
    attn_wmma<<<agrid, 128, 0, stream>>>(qb, kb, vT, aout);

    gemm_wmma<2><<<ggrid, 256, 0, stream>>>(aout, WpT, bp, d_out);
}
